// O3AttentionLayer_85358180041034
// MI455X (gfx1250) — compile-verified, matrix operand order checked by reading, not run on hardware
//
#include <hip/hip_runtime.h>
#include <hip/hip_bf16.h>
#include <math.h>

typedef __attribute__((ext_vector_type(16))) _Float16 v16h;
typedef __attribute__((ext_vector_type(8)))  float    v8f;

#define NN      20000      // nodes
#define NE      320000     // edges
#define NB      32         // basis
#define HIDN    16
#define KEYW    288        // 18 tiles of 16
#define VALW    576        // 36 tiles of 16
#define KT      18
#define VT      36
#define SKP     289        // padded LDS row stride (odd -> conflict-free)
#define SVP     577
#define SHP     17

// normalization constants (match reference)
#define S_W1    0.17677669529663689f   // 1/sqrt(32)
#define S_W2    0.25f                  // 1/sqrt(16)
#define A0N     0.17677669529663689f   // 1/sqrt(2*16)
#define A1N     0.25f                  // 1/sqrt(2*8)
#define A1S3    0.14433756729740643f   // A1N/sqrt(3)
#define SQ3     1.7320508075688772f
#define QN0     0.25f                  // 1/sqrt(16)
#define QN1     0.35355339059327373f   // 1/sqrt(8)
#define LN0     0.25f                  // 1/sqrt(2*8)
#define LN1     0.20412414523193154f   // 1/(sqrt(3)*sqrt(2*4))

// workspace layout (bytes, 256-aligned)
#define OFF_Q0    ((size_t)0)                    // N*8  f32
#define OFF_Q1    ((size_t)640000)               // N*12 f32
#define OFF_MAX   ((size_t)1600000)              // N f32
#define OFF_SUM   ((size_t)1680128)              // N f32
#define OFF_Z     ((size_t)1760256)              // E f32 (logits then z)
#define OFF_FW1K  ((size_t)3040256)              // 32*16 f16
#define OFF_FW2K  ((size_t)3041280)              // 18*32*16 f16
#define OFF_FW1V  ((size_t)3059712)              // 32*16 f16
#define OFF_FW2V  ((size_t)3060736)              // 36*32*16 f16

static __device__ __forceinline__ float frcp(float x) {
  return __builtin_amdgcn_rcpf(x);               // v_rcp_f32
}
static __device__ __forceinline__ float siluf(float x) {
  return x * frcp(1.f + __expf(-x));
}
// float atomic max via signed/unsigned int monotonicity trick
static __device__ __forceinline__ void atomicMaxF(float* a, float v) {
  if (v >= 0.f) atomicMax((int*)a, __float_as_int(v));
  else          atomicMin((unsigned int*)a, __float_as_uint(v));
}
// K index within an f16 A/B fragment: vgpr r (0..7), lane-half h
static __device__ __forceinline__ int kidx(int r, int h) {
  return ((r & 3) << 1) + (h << 3) + ((r >> 2) << 4);
}

// ---------------- init: zero output, -inf max, zero sum ----------------
__global__ void k_init(float* __restrict__ out, float* __restrict__ nmax,
                       float* __restrict__ nsum) {
  int i = blockIdx.x * blockDim.x + threadIdx.x;
  if (i < NN * 40) out[i] = 0.f;
  if (i < NN) { nmax[i] = -INFINITY; nsum[i] = 0.f; }
}

// ---------------- weight fragment prep (f32 -> f16 WMMA B-fragments) ----
__global__ void k_prep(const float* __restrict__ Wk1, const float* __restrict__ Wk2,
                       const float* __restrict__ Wv1, const float* __restrict__ Wv2,
                       _Float16* __restrict__ f1k, _Float16* __restrict__ f2k,
                       _Float16* __restrict__ f1v, _Float16* __restrict__ f2v) {
  int j = blockIdx.x * blockDim.x + threadIdx.x;   // 32 + 576 + 32 + 1152 jobs
  if (j >= 1792) return;
  if (j < 32) {                       // GEMM1 key B frag: K=32, N=16
    int lane = j, h = lane >> 4, n = lane & 15;
    _Float16* f = f1k + lane * 16;
#pragma unroll
    for (int r = 0; r < 8; ++r) {
      int k = kidx(r, h);
      f[2 * r]     = (_Float16)(Wk1[k * HIDN + n] * S_W1);
      f[2 * r + 1] = (_Float16)(Wk1[(k + 1) * HIDN + n] * S_W1);
    }
  } else if (j < 32 + KT * 32) {      // GEMM2 key B frags: K=16 (pad 32), 18 tiles
    int t = (j - 32) >> 5, lane = (j - 32) & 31, h = lane >> 4;
    int n = t * 16 + (lane & 15);
    _Float16* f = f2k + (t * 32 + lane) * 16;
#pragma unroll
    for (int r = 0; r < 8; ++r) {
      if (r < 4) {
        int k = (r << 1) + (h << 3);
        f[2 * r]     = (_Float16)(Wk2[k * KEYW + n] * S_W2);
        f[2 * r + 1] = (_Float16)(Wk2[(k + 1) * KEYW + n] * S_W2);
      } else { f[2 * r] = (_Float16)0.f; f[2 * r + 1] = (_Float16)0.f; }
    }
  } else if (j < 32 + KT * 32 + 32) { // GEMM1 value B frag
    int lane = j - (32 + KT * 32), h = lane >> 4, n = lane & 15;
    _Float16* f = f1v + lane * 16;
#pragma unroll
    for (int r = 0; r < 8; ++r) {
      int k = kidx(r, h);
      f[2 * r]     = (_Float16)(Wv1[k * HIDN + n] * S_W1);
      f[2 * r + 1] = (_Float16)(Wv1[(k + 1) * HIDN + n] * S_W1);
    }
  } else {                            // GEMM2 value B frags: 36 tiles
    int q = j - (64 + KT * 32);
    int t = q >> 5, lane = q & 31, h = lane >> 4;
    int n = t * 16 + (lane & 15);
    _Float16* f = f2v + (t * 32 + lane) * 16;
#pragma unroll
    for (int r = 0; r < 8; ++r) {
      if (r < 4) {
        int k = (r << 1) + (h << 3);
        f[2 * r]     = (_Float16)(Wv2[k * VALW + n] * S_W2);
        f[2 * r + 1] = (_Float16)(Wv2[(k + 1) * VALW + n] * S_W2);
      } else { f[2 * r] = (_Float16)0.f; f[2 * r + 1] = (_Float16)0.f; }
    }
  }
}

// ---------------- per-node query projection ----------------
__global__ void k_query(const float* __restrict__ x0, const float* __restrict__ x1,
                        const float* __restrict__ Wq0, const float* __restrict__ Wq1,
                        float* __restrict__ q0b, float* __restrict__ q1b) {
  int n = blockIdx.x * blockDim.x + threadIdx.x;
  if (n >= NN) return;
  float xv[16];
#pragma unroll
  for (int u = 0; u < 16; ++u) xv[u] = x0[n * 16 + u];
#pragma unroll
  for (int w = 0; w < 8; ++w) {
    float a = 0.f;
#pragma unroll
    for (int u = 0; u < 16; ++u) a += xv[u] * Wq0[u * 8 + w];
    q0b[n * 8 + w] = QN0 * a;
  }
#pragma unroll
  for (int w = 0; w < 4; ++w)
#pragma unroll
    for (int i = 0; i < 3; ++i) {
      float a = 0.f;
#pragma unroll
      for (int u = 0; u < 8; ++u) a += x1[n * 24 + u * 3 + i] * Wq1[u * 4 + w];
      q1b[n * 12 + w * 3 + i] = QN1 * a;
    }
}

// ---------------- pass A: keys + logits (WMMA radial MLP fused with TP) --
__global__ __launch_bounds__(64) void k_key(
    const float* __restrict__ x0, const float* __restrict__ x1,
    const float* __restrict__ pos, const float* __restrict__ eb,
    const int* __restrict__ src, const int* __restrict__ dst,
    const float* __restrict__ Ws00, const float* __restrict__ Ws11,
    const float* __restrict__ q0b, const float* __restrict__ q1b,
    const _Float16* __restrict__ f1, const _Float16* __restrict__ f2,
    float* __restrict__ zbuf, float* __restrict__ nmax) {
  __shared__ float sH[2][16][SHP];
  __shared__ float sW[2][16][SKP];
  const int wave = threadIdx.x >> 5, lane = threadIdx.x & 31;
  const int h = lane >> 4, ln = lane & 15;
  const int e0 = (blockIdx.x * 2 + wave) * 16;

  // GEMM1: basis[16x32] @ W1 -> H[16x16], silu
  v16h a1;
  {
    const float* br = eb + (size_t)(e0 + ln) * NB;   // A row = edge
#pragma unroll
    for (int r = 0; r < 8; ++r) {
      int k = kidx(r, h);
      a1[2 * r] = (_Float16)br[k]; a1[2 * r + 1] = (_Float16)br[k + 1];
    }
  }
  v16h b1 = *(const v16h*)(f1 + lane * 16);
  v8f hh = {};
  hh = __builtin_amdgcn_wmma_f32_16x16x32_f16(false, a1, false, b1, (short)0, hh, false, false);
#pragma unroll
  for (int r = 0; r < 8; ++r) sH[wave][r + 8 * h][ln] = siluf(hh[r]);
  __syncthreads();

  // A2 frag from H (K=16, upper half zero)
  v16h a2;
#pragma unroll
  for (int r = 0; r < 8; ++r) {
    if (r < 4) {
      int k = (r << 1) + (h << 3);
      a2[2 * r] = (_Float16)sH[wave][ln][k]; a2[2 * r + 1] = (_Float16)sH[wave][ln][k + 1];
    } else { a2[2 * r] = (_Float16)0.f; a2[2 * r + 1] = (_Float16)0.f; }
  }
  // GEMM2: H @ W2 -> wk[16x288] in LDS
#pragma unroll
  for (int t = 0; t < KT; ++t) {
    v16h b = *(const v16h*)(f2 + (t * 32 + lane) * 16);
    v8f d = {};
    d = __builtin_amdgcn_wmma_f32_16x16x32_f16(false, a2, false, b, (short)0, d, false, false);
#pragma unroll
    for (int r = 0; r < 8; ++r) sW[wave][r + 8 * h][t * 16 + ln] = d[r];
  }
  __syncthreads();

  // per-edge TP + logit: lane pair (ln, ln+16) shares edge ln; h picks half
  {
    const int e = e0 + ln;
    const float* wr = sW[wave][ln];
    const int s = src[e], dn = dst[e];
    float ev0 = pos[dn * 3 + 0] - pos[s * 3 + 0];
    float ev1 = pos[dn * 3 + 1] - pos[s * 3 + 1];
    float ev2 = pos[dn * 3 + 2] - pos[s * 3 + 2];
    float inv = SQ3 * frcp(__builtin_amdgcn_sqrtf(ev0 * ev0 + ev1 * ev1 + ev2 * ev2) + 1e-8f);
    float sh[3] = {ev0 * inv, ev1 * inv, ev2 * inv};
    float x0v[16], x1v[24], d1[8];
#pragma unroll
    for (int u = 0; u < 16; ++u) x0v[u] = x0[s * 16 + u];
#pragma unroll
    for (int u = 0; u < 24; ++u) x1v[u] = x1[s * 24 + u];
#pragma unroll
    for (int u = 0; u < 8; ++u)
      d1[u] = x1v[u * 3] * sh[0] + x1v[u * 3 + 1] * sh[1] + x1v[u * 3 + 2] * sh[2];
    float q0d[8], q1d[12];
#pragma unroll
    for (int u = 0; u < 8; ++u) q0d[u] = q0b[dn * 8 + u];
#pragma unroll
    for (int u = 0; u < 12; ++u) q1d[u] = q1b[dn * 12 + u];

    // k0 half: v = 4h..4h+3, and partial l0
    float l0 = 0.f;
#pragma unroll
    for (int w = 0; w < 4; ++w) {
      const int v = 4 * h + w;
      float a = 0.f, b = 0.f;
#pragma unroll
      for (int u = 0; u < 16; ++u) a += x0v[u] * wr[u * 8 + v];
#pragma unroll
      for (int u = 0; u < 8; ++u) b += d1[u] * wr[128 + u * 8 + v];
      float k0v = A0N * a + A1S3 * b;
      float t = 0.f;
#pragma unroll
      for (int u = 0; u < 8; ++u) t += q0d[u] * Ws00[u * 8 + v];
      l0 += t * k0v;
    }
    // k1 half: w = 2h, 2h+1, and partial l1
    float l1 = 0.f;
#pragma unroll
    for (int j = 0; j < 2; ++j) {
      const int w = 2 * h + j;
      float p = 0.f;
#pragma unroll
      for (int u = 0; u < 16; ++u) p += x0v[u] * wr[192 + u * 4 + w];
      float k1v[3];
#pragma unroll
      for (int i = 0; i < 3; ++i) {
        float vx = 0.f;
#pragma unroll
        for (int u = 0; u < 8; ++u) vx += x1v[u * 3 + i] * wr[256 + u * 4 + w];
        k1v[i] = A0N * p * sh[i] + A1N * vx;
      }
#pragma unroll
      for (int u = 0; u < 4; ++u) {
        float dz = q1d[u * 3 + 0] * k1v[0] + q1d[u * 3 + 1] * k1v[1]
                 + q1d[u * 3 + 2] * k1v[2];
        l1 += dz * Ws11[u * 4 + w];
      }
    }
    float part = LN0 * l0 + LN1 * l1;
    float lg = part + __shfl_xor(part, 16, 32);
    if (h == 0) {
      zbuf[e] = lg;
      atomicMaxF(nmax + dn, lg);
    }
  }
}

// ---------------- pass B: exp + segment sum ----------------
__global__ void k_soft(const int* __restrict__ dst, const float* __restrict__ nmax,
                       float* __restrict__ zbuf, float* __restrict__ nsum) {
  int e = blockIdx.x * blockDim.x + threadIdx.x;
  if (e >= NE) return;
  int dn = dst[e];
  float z = __expf(zbuf[e] - nmax[dn]);
  zbuf[e] = z;
  atomicAdd(nsum + dn, z);
}

// ---------------- pass C: values (WMMA) + weighted scatter ----------------
__global__ __launch_bounds__(32) void k_val(
    const float* __restrict__ x0, const float* __restrict__ x1,
    const float* __restrict__ pos, const float* __restrict__ eb,
    const int* __restrict__ src, const int* __restrict__ dst,
    const _Float16* __restrict__ f1, const _Float16* __restrict__ f2,
    const float* __restrict__ zbuf, const float* __restrict__ nsum,
    float* __restrict__ out) {
  __shared__ float sH[16][SHP];
  __shared__ float sV[16][SVP];
  const int lane = threadIdx.x & 31, h = lane >> 4, ln = lane & 15;
  const int e0 = blockIdx.x * 16;

  v16h a1;
  {
    const float* br = eb + (size_t)(e0 + ln) * NB;
#pragma unroll
    for (int r = 0; r < 8; ++r) {
      int k = kidx(r, h);
      a1[2 * r] = (_Float16)br[k]; a1[2 * r + 1] = (_Float16)br[k + 1];
    }
  }
  v16h b1 = *(const v16h*)(f1 + lane * 16);
  v8f hh = {};
  hh = __builtin_amdgcn_wmma_f32_16x16x32_f16(false, a1, false, b1, (short)0, hh, false, false);
#pragma unroll
  for (int r = 0; r < 8; ++r) sH[r + 8 * h][ln] = siluf(hh[r]);
  __syncthreads();

  v16h a2;
#pragma unroll
  for (int r = 0; r < 8; ++r) {
    if (r < 4) {
      int k = (r << 1) + (h << 3);
      a2[2 * r] = (_Float16)sH[ln][k]; a2[2 * r + 1] = (_Float16)sH[ln][k + 1];
    } else { a2[2 * r] = (_Float16)0.f; a2[2 * r + 1] = (_Float16)0.f; }
  }
#pragma unroll
  for (int t = 0; t < VT; ++t) {
    v16h b = *(const v16h*)(f2 + (t * 32 + lane) * 16);
    v8f d = {};
    d = __builtin_amdgcn_wmma_f32_16x16x32_f16(false, a2, false, b, (short)0, d, false, false);
#pragma unroll
    for (int r = 0; r < 8; ++r) sV[r + 8 * h][t * 16 + ln] = d[r];
  }
  __syncthreads();

  // per-edge: lane pair (ln, ln+16) shares edge ln; h picks output half
  {
    const int e = e0 + ln;
    const float* wr = sV[ln];
    const int s = src[e], dn = dst[e];
    float ev0 = pos[dn * 3 + 0] - pos[s * 3 + 0];
    float ev1 = pos[dn * 3 + 1] - pos[s * 3 + 1];
    float ev2 = pos[dn * 3 + 2] - pos[s * 3 + 2];
    float inv = SQ3 * frcp(__builtin_amdgcn_sqrtf(ev0 * ev0 + ev1 * ev1 + ev2 * ev2) + 1e-8f);
    float sh[3] = {ev0 * inv, ev1 * inv, ev2 * inv};
    float x0v[16], x1v[24], d1[8];
#pragma unroll
    for (int u = 0; u < 16; ++u) x0v[u] = x0[s * 16 + u];
#pragma unroll
    for (int u = 0; u < 24; ++u) x1v[u] = x1[s * 24 + u];
#pragma unroll
    for (int u = 0; u < 8; ++u)
      d1[u] = x1v[u * 3] * sh[0] + x1v[u * 3 + 1] * sh[1] + x1v[u * 3 + 2] * sh[2];
    const float al = zbuf[e] * frcp(nsum[dn]);
    float* ob = out + dn * 40;
    // v0 half: w = 8h..8h+7
#pragma unroll
    for (int w = 0; w < 8; ++w) {
      const int wo = 8 * h + w;
      float a = 0.f, b = 0.f;
#pragma unroll
      for (int u = 0; u < 16; ++u) a += x0v[u] * wr[u * 16 + wo];
#pragma unroll
      for (int u = 0; u < 8; ++u) b += d1[u] * wr[256 + u * 16 + wo];
      atomicAdd(ob + wo, al * (A0N * a + A1S3 * b));
    }
    // v1 half: w = 4h..4h+3
#pragma unroll
    for (int j = 0; j < 4; ++j) {
      const int w = 4 * h + j;
      float p = 0.f;
#pragma unroll
      for (int u = 0; u < 16; ++u) p += x0v[u] * wr[384 + u * 8 + w];
#pragma unroll
      for (int i = 0; i < 3; ++i) {
        float vx = 0.f;
#pragma unroll
        for (int u = 0; u < 8; ++u) vx += x1v[u * 3 + i] * wr[512 + u * 8 + w];
        atomicAdd(ob + 16 + w * 3 + i, al * (A0N * p * sh[i] + A1N * vx));
      }
    }
  }
}

extern "C" void kernel_launch(void* const* d_in, const int* in_sizes, int n_in,
                              void* d_out, int out_size, void* d_ws, size_t ws_size,
                              hipStream_t stream) {
  const float* x0   = (const float*)d_in[0];
  const float* x1   = (const float*)d_in[1];
  const float* pos  = (const float*)d_in[2];
  const float* eb   = (const float*)d_in[3];
  const int*   src  = (const int*)d_in[4];
  const int*   dst  = (const int*)d_in[5];
  const float* Wq0  = (const float*)d_in[6];
  const float* Wq1  = (const float*)d_in[7];
  const float* Wk1  = (const float*)d_in[8];
  const float* Wk2  = (const float*)d_in[9];
  const float* Wv1  = (const float*)d_in[10];
  const float* Wv2  = (const float*)d_in[11];
  const float* Ws00 = (const float*)d_in[12];
  const float* Ws11 = (const float*)d_in[13];
  float* out = (float*)d_out;

  char* ws = (char*)d_ws;
  float*    q0b  = (float*)(ws + OFF_Q0);
  float*    q1b  = (float*)(ws + OFF_Q1);
  float*    nmax = (float*)(ws + OFF_MAX);
  float*    nsum = (float*)(ws + OFF_SUM);
  float*    zbuf = (float*)(ws + OFF_Z);
  _Float16* f1k  = (_Float16*)(ws + OFF_FW1K);
  _Float16* f2k  = (_Float16*)(ws + OFF_FW2K);
  _Float16* f1v  = (_Float16*)(ws + OFF_FW1V);
  _Float16* f2v  = (_Float16*)(ws + OFF_FW2V);

  k_init<<<(NN * 40 + 255) / 256, 256, 0, stream>>>(out, nmax, nsum);
  k_prep<<<7, 256, 0, stream>>>(Wk1, Wk2, Wv1, Wv2, f1k, f2k, f1v, f2v);
  k_query<<<(NN + 255) / 256, 256, 0, stream>>>(x0, x1, Wq0, Wq1, q0b, q1b);
  k_key<<<NE / 32, 64, 0, stream>>>(x0, x1, pos, eb, src, dst, Ws00, Ws11,
                                    q0b, q1b, f1k, f2k, zbuf, nmax);
  k_soft<<<(NE + 255) / 256, 256, 0, stream>>>(dst, nmax, zbuf, nsum);
  k_val<<<NE / 16, 32, 0, stream>>>(x0, x1, pos, eb, src, dst, f1v, f2v,
                                    zbuf, nsum, out);
}
